// ST_Block_28295244546604
// MI455X (gfx1250) — matrix operand order repaired and checked by reading, not compiled, not run on hardware
//
#include <hip/hip_runtime.h>
#include <math.h>

typedef __attribute__((ext_vector_type(8)))  __bf16 v8bf;
typedef __attribute__((ext_vector_type(16))) __bf16 v16bf;
typedef __attribute__((ext_vector_type(8)))  float  v8f;

#define N_NODES 10000
#define N_EDGES 160000
#define TS_DIM  10
#define F_DIM   640
#define TS2_DIM 8
#define M1_ROWS 100000   // N * TS
#define M2_ROWS 80000    // N * TS2  (80000 = 625*128, already tile-aligned)
#define K_TCN   192      // 64 channels * kernel 3
#define M1_PAD  100096   // round_up(M1_ROWS, 128)
#define NG_PAD  10112    // round_up(N_NODES, 128)

__device__ __forceinline__ unsigned short f2bf(float f) {
  unsigned int u = __float_as_uint(f);
  unsigned int r = u + 0x7FFFu + ((u >> 16) & 1u);   // round-to-nearest-even
  return (unsigned short)(r >> 16);
}
__device__ __forceinline__ v16bf cat16(v8bf lo, v8bf hi) {
  return __builtin_shufflevector(lo, hi, 0,1,2,3,4,5,6,7,8,9,10,11,12,13,14,15);
}
__device__ __forceinline__ v8f f8_zero() {
  v8f z = {0.f,0.f,0.f,0.f,0.f,0.f,0.f,0.f};
  return z;
}
__device__ __forceinline__ float sigmf(float x) { return 1.0f / (1.0f + expf(-x)); }

// ---------------------------------------------------------------------------
// bf16 WMMA GEMM: C[M,Nc] (f32) = A[M,K] (bf16, lda) x B[K,Nc], B stored
// transposed Bt[Nc][K] (ldb = K).
// Block tile 128x128, 8 waves; wave w computes rows 16w..16w+15 x all 128 cols
// (eight 16x16 f32 accumulators), K stepped by 32.
// A and Bt must be padded/readable for full 128-row / 128-col tiles (pad rows
// and pad cols zero-filled by the host); M/Nc only guard the C stores.
// Staging uses CDNA5 async global->LDS B128 copies (ASYNCcnt) when available.
// All 8 B fragments are loaded into independent registers before the WMMA
// burst so the ds_loads clause under one s_wait_dscnt and the eight
// v_wmma_f32_16x16x32_bf16 issue back-to-back.
// ---------------------------------------------------------------------------
__global__ __launch_bounds__(256) void gemm_bf16_wmma(
    const unsigned short* __restrict__ A, int lda,
    const unsigned short* __restrict__ Bt, int ldb,
    float* __restrict__ Cmat, int ldc,
    int M, int Nc, int K)
{
  __shared__ __align__(16) unsigned short As[128][32];   // 8 KB
  __shared__ __align__(16) unsigned short Bs[128][32];   // 8 KB

  const int tid  = threadIdx.x;
  const int lane = tid & 31;
  const int wave = tid >> 5;                  // 8 row-waves, 16 rows each
  const int rowBlock = blockIdx.y * 128;
  const int colBlock = blockIdx.x * 128;

  v8f acc[8];
#pragma unroll
  for (int t = 0; t < 8; ++t) acc[t] = f8_zero();

  // Staging map: 512 16B chunks per tile; chunk q -> row q>>2, elem (q&3)*8.
  // Thread stages chunks {tid, tid+256} of As and of Bs.
  const int rS = tid >> 2;             // 0..63
  const int jS = (tid & 3) * 8;        // 0,8,16,24
  const unsigned short* pA0 = A  + (size_t)(rowBlock + rS)      * lda + jS;
  const unsigned short* pA1 = A  + (size_t)(rowBlock + rS + 64) * lda + jS;
  const unsigned short* pB0 = Bt + (size_t)(colBlock + rS)      * ldb + jS;
  const unsigned short* pB1 = Bt + (size_t)(colBlock + rS + 64) * ldb + jS;
#if defined(__gfx1250__)
  const unsigned oA0 = (unsigned)(size_t)&As[0][0] + (unsigned)tid * 16u;
  const unsigned oA1 = oA0 + 4096u;
  const unsigned oB0 = (unsigned)(size_t)&Bs[0][0] + (unsigned)tid * 16u;
  const unsigned oB1 = oB0 + 4096u;
#endif

  for (int k0 = 0; k0 < K; k0 += 32) {
#if defined(__gfx1250__)
    // Async DMA global -> LDS (tracked by ASYNCcnt, no VGPR staging)
    asm volatile("global_load_async_to_lds_b128 %0, %1, off"
                 :: "v"(oA0), "v"(pA0 + k0) : "memory");
    asm volatile("global_load_async_to_lds_b128 %0, %1, off"
                 :: "v"(oA1), "v"(pA1 + k0) : "memory");
    asm volatile("global_load_async_to_lds_b128 %0, %1, off"
                 :: "v"(oB0), "v"(pB0 + k0) : "memory");
    asm volatile("global_load_async_to_lds_b128 %0, %1, off"
                 :: "v"(oB1), "v"(pB1 + k0) : "memory");
    asm volatile("s_wait_asynccnt 0x0" ::: "memory");
#else
    {
      v8bf a0 = *(const v8bf*)(pA0 + k0);
      v8bf a1 = *(const v8bf*)(pA1 + k0);
      v8bf b0 = *(const v8bf*)(pB0 + k0);
      v8bf b1 = *(const v8bf*)(pB1 + k0);
      *(v8bf*)&As[rS][jS]      = a0;
      *(v8bf*)&As[rS + 64][jS] = a1;
      *(v8bf*)&Bs[rS][jS]      = b0;
      *(v8bf*)&Bs[rS + 64][jS] = b1;
    }
#endif
    __syncthreads();

    // A fragment (16-bit A 16x32 layout):
    // lanes 0-15: row=lane, K {0..7,16..23}; lanes 16-31: row=lane-16, K {8..15,24..31}
    const int am = wave * 16 + (lane & 15);
    const int ak = (lane < 16) ? 0 : 8;
    v8bf alo = *(const v8bf*)&As[am][ak];
    v8bf ahi = *(const v8bf*)&As[am][ak + 16];
    v16bf afrag = cat16(alo, ahi);

    // Load all 8 B fragments into independent registers first (one ds clause),
    // then issue the 8 WMMAs back-to-back (no D->A/B hazard between them).
    v16bf bfrag[8];
    {
      const int bn0 = lane & 15;
      const int bk  = (lane < 16) ? 0 : 8;
#pragma unroll
      for (int t = 0; t < 8; ++t) {
        v8bf blo = *(const v8bf*)&Bs[t * 16 + bn0][bk];
        v8bf bhi = *(const v8bf*)&Bs[t * 16 + bn0][bk + 16];
        bfrag[t] = cat16(blo, bhi);
      }
    }
#pragma unroll
    for (int t = 0; t < 8; ++t) {
      acc[t] = __builtin_amdgcn_wmma_f32_16x16x32_bf16(
          false, afrag, false, bfrag[t], (short)0, acc[t], false, false);
    }
    __syncthreads();
  }

  // C/D layout: VGPR v -> M=v (lanes 0-15) / M=v+8 (lanes 16-31), N = lane%16
  const int mBase = rowBlock + wave * 16 + ((lane < 16) ? 0 : 8);
  const int nLane = lane & 15;
#pragma unroll
  for (int t = 0; t < 8; ++t) {
    int col = colBlock + t * 16 + nLane;
    if (col >= Nc) continue;
#pragma unroll
    for (int v = 0; v < 8; ++v) {
      int row = mBase + v;
      if (row < M) Cmat[(size_t)row * ldc + col] = acc[t][v];
    }
  }
}

// ---------------------------------------------------------------------------
// Elementwise / packing / graph kernels
// ---------------------------------------------------------------------------
__global__ void k_fill(float* __restrict__ p, float v, int n) {
  int i = blockIdx.x * blockDim.x + threadIdx.x;
  if (i < n) p[i] = v;
}
__global__ void k_fill_u16(unsigned short* __restrict__ p, unsigned short v, int n) {
  int i = blockIdx.x * blockDim.x + threadIdx.x;
  if (i < n) p[i] = v;
}

// Pack two TCN weight tensors [64,64,3] into bf16 Bt[128][192], k = c*3+kk
__global__ void k_pack_tcnw(const float* __restrict__ w1, const float* __restrict__ w2,
                            unsigned short* __restrict__ out) {
  int idx = blockIdx.x * blockDim.x + threadIdx.x;
  if (idx >= 128 * K_TCN) return;
  int j = idx / K_TCN, k = idx % K_TCN;
  int c = k / 3, kk = k % 3;
  float v = (j < 64) ? w1[(j * 64 + c) * 3 + kk] : w2[((j - 64) * 64 + c) * 3 + kk];
  out[idx] = f2bf(v);
}

// Transpose-pack W[K][Nc] f32 -> Bt[Nc][K] bf16
__global__ void k_pack_T(const float* __restrict__ W, unsigned short* __restrict__ out,
                         int K, int Nc) {
  int idx = blockIdx.x * blockDim.x + threadIdx.x;
  if (idx >= K * Nc) return;
  int j = idx / K, k = idx % K;
  out[idx] = f2bf(W[(size_t)k * Nc + j]);
}

// im2col for TCN1: A1[(n*10+t)][c*3+kk] = x[n][c][t+kk]
__global__ void k_im2col1(const float* __restrict__ x, unsigned short* __restrict__ A) {
  int idx = blockIdx.x * blockDim.x + threadIdx.x;
  if (idx >= M1_ROWS * K_TCN) return;
  int row = idx / K_TCN, k = idx % K_TCN;
  int n = row / TS_DIM, t = row % TS_DIM;
  int c = k / 3, kk = k % 3;
  A[idx] = f2bf(x[((size_t)n * 64 + c) * 12 + t + kk]);
}

// Gated activation of TCN1 -> out0 bf16 in [N][h*10+t] layout
__global__ void k_gated1(const float* __restrict__ G, const float* __restrict__ b1,
                         const float* __restrict__ b2, unsigned short* __restrict__ out0b) {
  int idx = blockIdx.x * blockDim.x + threadIdx.x;
  if (idx >= M1_ROWS * 64) return;
  int row = idx >> 6, j = idx & 63;
  int n = row / TS_DIM, t = row % TS_DIM;
  float a = G[(size_t)row * 128 + j] + b1[j];
  float b = G[(size_t)row * 128 + 64 + j] + b2[j];
  float v = tanhf(a) * sigmf(b);
  out0b[(size_t)n * F_DIM + j * TS_DIM + t] = f2bf(v);
}

// psi MLP: relu(ND_t @ W1 + b1) @ W2 + b2  -> psi[N][3]
__global__ void k_psi(const float* __restrict__ NDt, const float* __restrict__ W1,
                      const float* __restrict__ b1, const float* __restrict__ W2,
                      const float* __restrict__ b2, float* __restrict__ psi) {
  int n = blockIdx.x * blockDim.x + threadIdx.x;
  if (n >= N_NODES) return;
  float nd0 = NDt[n * 3 + 0], nd1 = NDt[n * 3 + 1], nd2 = NDt[n * 3 + 2];
  float a0 = b2[0], a1 = b2[1], a2 = b2[2];
  for (int j = 0; j < 64; ++j) {
    float h = fmaxf(nd0 * W1[j] + nd1 * W1[64 + j] + nd2 * W1[128 + j] + b1[j], 0.f);
    a0 += h * W2[j * 3 + 0];
    a1 += h * W2[j * 3 + 1];
    a2 += h * W2[j * 3 + 2];
  }
  psi[n * 3 + 0] = a0; psi[n * 3 + 1] = a1; psi[n * 3 + 2] = a2;
}

__global__ void k_edgedeg(const int* __restrict__ ei, const float* __restrict__ ea,
                          int col, float* __restrict__ deg) {
  int e = blockIdx.x * blockDim.x + threadIdx.x;
  if (e >= N_EDGES) return;
  atomicAdd(&deg[ei[N_EDGES + e]], ea[(size_t)e * 2 + col] + 1.0f);
}

__global__ void k_dinv(const float* __restrict__ deg, float* __restrict__ dinv, int n) {
  int i = blockIdx.x * blockDim.x + threadIdx.x;
  if (i >= n) return;
  float d = deg[i];
  dinv[i] = (d > 0.f) ? rsqrtf(d) : 0.f;
}

// Self-loop term: gout = dinv[n]^2 * hlin  (overwrites gout fully)
__global__ void k_selfinit(const float* __restrict__ hlin, const float* __restrict__ dinv,
                           float* __restrict__ gout) {
  int idx = blockIdx.x * blockDim.x + threadIdx.x;
  if (idx >= N_NODES * F_DIM) return;
  int n = idx / F_DIM;
  float g = dinv[n];
  gout[idx] = g * g * hlin[idx];
}

// Edge scatter: gout[d] += dinv[s]*w*dinv[d] * hlin[s], 4 floats per thread
__global__ void k_scatter(const int* __restrict__ ei, const float* __restrict__ ea, int col,
                          const float* __restrict__ dinv, const float* __restrict__ hlin,
                          float* __restrict__ gout) {
  long long idx = (long long)blockIdx.x * blockDim.x + threadIdx.x;
  const long long total = (long long)N_EDGES * (F_DIM / 4);
  if (idx >= total) return;
  int e  = (int)(idx / (F_DIM / 4));
  int fq = (int)(idx % (F_DIM / 4)) * 4;
  int s = ei[e];
  int d = ei[N_EDGES + e];
  float w = ea[(size_t)e * 2 + col] + 1.0f;
  float nrm = dinv[s] * w * dinv[d];
  const float4 h4 = *(const float4*)(hlin + (size_t)s * F_DIM + fq);
  float* gp = gout + (size_t)d * F_DIM + fq;
  atomicAdd(gp + 0, nrm * h4.x);
  atomicAdd(gp + 1, nrm * h4.y);
  atomicAdd(gp + 2, nrm * h4.z);
  atomicAdd(gp + 3, nrm * h4.w);
}

// h = relu(gout + b); acc += psi[n][hop]*h; cur = bf16(h)
__global__ void k_hopfinish(const float* __restrict__ gout, const float* __restrict__ bias,
                            const float* __restrict__ psi, int hop,
                            float* __restrict__ acc, unsigned short* __restrict__ cur) {
  int idx = blockIdx.x * blockDim.x + threadIdx.x;
  if (idx >= N_NODES * F_DIM) return;
  int n = idx / F_DIM, f = idx % F_DIM;
  float h = fmaxf(gout[idx] + bias[f], 0.f);
  acc[idx] += psi[n * 3 + hop] * h;
  cur[idx] = f2bf(h);
}

__global__ void k_negfinish(const float* __restrict__ gout, const float* __restrict__ bias,
                            float* __restrict__ outneg) {
  int idx = blockIdx.x * blockDim.x + threadIdx.x;
  if (idx >= N_NODES * F_DIM) return;
  int f = idx % F_DIM;
  outneg[idx] = fmaxf(gout[idx] + bias[f], 0.f);
}

// cat[(n*10+t)][k] = k<64 ? acc[n][k*10+t] : outneg[n][(k-64)*10+t]   (bf16)
__global__ void k_build_cat(const float* __restrict__ acc, const float* __restrict__ outneg,
                            unsigned short* __restrict__ catb) {
  int idx = blockIdx.x * blockDim.x + threadIdx.x;
  if (idx >= M1_ROWS * 128) return;
  int row = idx >> 7, k = idx & 127;
  int n = row / TS_DIM, t = row % TS_DIM;
  float v = (k < 64) ? acc[(size_t)n * F_DIM + k * TS_DIM + t]
                     : outneg[(size_t)n * F_DIM + (k - 64) * TS_DIM + t];
  catb[idx] = f2bf(v);
}

// out2[n][d][t] = relu(out2r[(n*10+t)][d] + gre_b[d])
__global__ void k_relu_out2(const float* __restrict__ out2r, const float* __restrict__ greb,
                            float* __restrict__ out2buf) {
  int idx = blockIdx.x * blockDim.x + threadIdx.x;
  if (idx >= M1_ROWS * 64) return;
  int row = idx >> 6, d = idx & 63;
  int n = row / TS_DIM, t = row % TS_DIM;
  out2buf[((size_t)n * 64 + d) * TS_DIM + t] = fmaxf(out2r[(size_t)row * 64 + d] + greb[d], 0.f);
}

// im2col for TCN2: A2[(n*8+t2)][d*3+kk] = out2[n][d][t2+kk]
__global__ void k_im2col2(const float* __restrict__ out2buf, unsigned short* __restrict__ A) {
  int idx = blockIdx.x * blockDim.x + threadIdx.x;
  if (idx >= M2_ROWS * K_TCN) return;
  int row = idx / K_TCN, k = idx % K_TCN;
  int n = row / TS2_DIM, t2 = row % TS2_DIM;
  int d = k / 3, kk = k % 3;
  A[idx] = f2bf(out2buf[((size_t)n * 64 + d) * TS_DIM + t2 + kk]);
}

// Final gated activation -> d_out [N,64,8] f32
__global__ void k_gated2(const float* __restrict__ G, const float* __restrict__ b1,
                         const float* __restrict__ b2, float* __restrict__ out) {
  int idx = blockIdx.x * blockDim.x + threadIdx.x;
  if (idx >= M2_ROWS * 64) return;
  int row = idx >> 6, j = idx & 63;
  int n = row / TS2_DIM, t2 = row % TS2_DIM;
  float a = G[(size_t)row * 128 + j] + b1[j];
  float b = G[(size_t)row * 128 + 64 + j] + b2[j];
  out[((size_t)n * 64 + j) * TS2_DIM + t2] = tanhf(a) * sigmf(b);
}

// ---------------------------------------------------------------------------
extern "C" void kernel_launch(void* const* d_in, const int* in_sizes, int n_in,
                              void* d_out, int out_size, void* d_ws, size_t ws_size,
                              hipStream_t stream) {
  (void)in_sizes; (void)n_in; (void)out_size; (void)ws_size;
  const float* x      = (const float*)d_in[0];
  const int*   ei     = (const int*)  d_in[1];
  const float* ea     = (const float*)d_in[2];
  const float* NDt    = (const float*)d_in[3];
  const float* tc1_w1 = (const float*)d_in[4];
  const float* tc1_b1 = (const float*)d_in[5];
  const float* tc1_w2 = (const float*)d_in[6];
  const float* tc1_b2 = (const float*)d_in[7];
  const float* W_pos  = (const float*)d_in[8];
  const float* b_pos  = (const float*)d_in[9];
  const float* W_neg  = (const float*)d_in[10];
  const float* b_neg  = (const float*)d_in[11];
  const float* gre_W  = (const float*)d_in[12];
  const float* gre_b  = (const float*)d_in[13];
  const float* psi_W1 = (const float*)d_in[14];
  const float* psi_b1 = (const float*)d_in[15];
  const float* psi_W2 = (const float*)d_in[16];
  const float* psi_b2 = (const float*)d_in[17];
  const float* tc2_w1 = (const float*)d_in[18];
  const float* tc2_b1 = (const float*)d_in[19];
  const float* tc2_w2 = (const float*)d_in[20];
  const float* tc2_b2 = (const float*)d_in[21];
  float* out = (float*)d_out;

  // ---- workspace bump allocation (aliased across pipeline phases) ----
  char* ws = (char*)d_ws;
  size_t off = 0;
  auto take = [&](size_t bytes) {
    size_t o = off; off += (bytes + 255) & ~(size_t)255; return o;
  };
  unsigned short* A1    = (unsigned short*)(ws + take((size_t)M1_PAD * K_TCN * 2)); // im2col; reused for TCN2
  float*          Gbuf  = (float*)(ws + take((size_t)M1_ROWS * 128 * 4));  // G1 -> cat(bf16, padded) -> G2
  unsigned short* out0b = (unsigned short*)(ws + take((size_t)NG_PAD * F_DIM * 2));
  float*          hlin  = (float*)(ws + take((size_t)N_NODES * F_DIM * 4)); // also out2 buffer
  float*          gout  = (float*)(ws + take((size_t)N_NODES * F_DIM * 4)); // also GReTo GEMM out
  unsigned short* cur   = (unsigned short*)(ws + take((size_t)NG_PAD * F_DIM * 2));
  float*          accb  = (float*)(ws + take((size_t)N_NODES * F_DIM * 4));
  float*          negb  = (float*)(ws + take((size_t)N_NODES * F_DIM * 4));
  unsigned short* wcat1 = (unsigned short*)(ws + take(128 * K_TCN * 2));
  unsigned short* wposT = (unsigned short*)(ws + take((size_t)F_DIM * F_DIM * 2));
  unsigned short* wnegT = (unsigned short*)(ws + take((size_t)F_DIM * F_DIM * 2));
  unsigned short* wgreT = (unsigned short*)(ws + take(128 * 128 * 2));  // 64 real + 64 pad cols
  unsigned short* wcat2 = (unsigned short*)(ws + take(128 * K_TCN * 2));
  float*          degb  = (float*)(ws + take((size_t)N_NODES * 4));
  float*          dinvP = (float*)(ws + take((size_t)N_NODES * 4));
  float*          dinvN = (float*)(ws + take((size_t)N_NODES * 4));
  float*          psib  = (float*)(ws + take((size_t)N_NODES * 3 * 4));

  auto blocks = [](long long n) { return dim3((unsigned)((n + 255) / 256)); };
  dim3 tpb(256);

  // ---- zero-fill pad tails so async 128-tile staging never reads garbage ----
  k_fill_u16<<<blocks((M1_PAD - M1_ROWS) * K_TCN), tpb, 0, stream>>>(
      A1 + (size_t)M1_ROWS * K_TCN, 0, (M1_PAD - M1_ROWS) * K_TCN);
  k_fill_u16<<<blocks((NG_PAD - N_NODES) * F_DIM), tpb, 0, stream>>>(
      out0b + (size_t)N_NODES * F_DIM, 0, (NG_PAD - N_NODES) * F_DIM);
  k_fill_u16<<<blocks((NG_PAD - N_NODES) * F_DIM), tpb, 0, stream>>>(
      cur + (size_t)N_NODES * F_DIM, 0, (NG_PAD - N_NODES) * F_DIM);
  k_fill_u16<<<blocks(128 * 128), tpb, 0, stream>>>(wgreT, 0, 128 * 128); // pack fills rows 0..63

  // ---- weight packing (bf16, transposed for GEMM B operand) ----
  k_pack_tcnw<<<blocks(128 * K_TCN), tpb, 0, stream>>>(tc1_w1, tc1_w2, wcat1);
  k_pack_tcnw<<<blocks(128 * K_TCN), tpb, 0, stream>>>(tc2_w1, tc2_w2, wcat2);
  k_pack_T<<<blocks((long long)F_DIM * F_DIM), tpb, 0, stream>>>(W_pos, wposT, F_DIM, F_DIM);
  k_pack_T<<<blocks((long long)F_DIM * F_DIM), tpb, 0, stream>>>(W_neg, wnegT, F_DIM, F_DIM);
  k_pack_T<<<blocks(128 * 64), tpb, 0, stream>>>(gre_W, wgreT, 128, 64);

  // ---- psi MLP + degree / norm precompute ----
  k_psi<<<blocks(N_NODES), tpb, 0, stream>>>(NDt, psi_W1, psi_b1, psi_W2, psi_b2, psib);
  k_fill<<<blocks(N_NODES), tpb, 0, stream>>>(degb, 1.0f, N_NODES);  // self-loop weight
  k_edgedeg<<<blocks(N_EDGES), tpb, 0, stream>>>(ei, ea, 0, degb);
  k_dinv<<<blocks(N_NODES), tpb, 0, stream>>>(degb, dinvP, N_NODES);
  k_fill<<<blocks(N_NODES), tpb, 0, stream>>>(degb, 1.0f, N_NODES);
  k_edgedeg<<<blocks(N_EDGES), tpb, 0, stream>>>(ei, ea, 1, degb);
  k_dinv<<<blocks(N_NODES), tpb, 0, stream>>>(degb, dinvN, N_NODES);

  // ---- TCN1: im2col + WMMA GEMM (M1 x 192 x 128) + gated activation ----
  k_im2col1<<<blocks((long long)M1_ROWS * K_TCN), tpb, 0, stream>>>(x, A1);
  {
    dim3 g(1, M1_PAD / 128);
    gemm_bf16_wmma<<<g, tpb, 0, stream>>>(A1, K_TCN, wcat1, K_TCN, Gbuf, 128,
                                          M1_ROWS, 128, K_TCN);
  }
  k_gated1<<<blocks((long long)M1_ROWS * 64), tpb, 0, stream>>>(Gbuf, tc1_b1, tc1_b2, out0b);

  // ---- 3 positive GCN hops ----
  k_fill<<<blocks((long long)N_NODES * F_DIM), tpb, 0, stream>>>(accb, 0.f, N_NODES * F_DIM);
  const unsigned short* hopIn = out0b;
  for (int hop = 0; hop < 3; ++hop) {
    dim3 g(F_DIM / 128, NG_PAD / 128);
    gemm_bf16_wmma<<<g, tpb, 0, stream>>>(hopIn, F_DIM, wposT, F_DIM, hlin, F_DIM,
                                          N_NODES, F_DIM, F_DIM);
    k_selfinit<<<blocks((long long)N_NODES * F_DIM), tpb, 0, stream>>>(hlin, dinvP, gout);
    k_scatter<<<blocks((long long)N_EDGES * (F_DIM / 4)), tpb, 0, stream>>>(
        ei, ea, 0, dinvP, hlin, gout);
    k_hopfinish<<<blocks((long long)N_NODES * F_DIM), tpb, 0, stream>>>(
        gout, b_pos, psib, hop, accb, cur);
    hopIn = cur;
  }

  // ---- negative-graph GCN ----
  {
    dim3 g(F_DIM / 128, NG_PAD / 128);
    gemm_bf16_wmma<<<g, tpb, 0, stream>>>(out0b, F_DIM, wnegT, F_DIM, hlin, F_DIM,
                                          N_NODES, F_DIM, F_DIM);
  }
  k_selfinit<<<blocks((long long)N_NODES * F_DIM), tpb, 0, stream>>>(hlin, dinvN, gout);
  k_scatter<<<blocks((long long)N_EDGES * (F_DIM / 4)), tpb, 0, stream>>>(
      ei, ea, 1, dinvN, hlin, gout);
  k_negfinish<<<blocks((long long)N_NODES * F_DIM), tpb, 0, stream>>>(gout, b_neg, negb);

  // ---- GReTo channel mix: concat -> WMMA GEMM (M1 x 128 x 64) -> relu+bias ----
  unsigned short* catb = (unsigned short*)Gbuf;   // Gbuf (G1) dead, reuse as bf16 cat (padded rows)
  k_fill_u16<<<blocks((M1_PAD - M1_ROWS) * 128), tpb, 0, stream>>>(
      catb + (size_t)M1_ROWS * 128, 0, (M1_PAD - M1_ROWS) * 128);
  k_build_cat<<<blocks((long long)M1_ROWS * 128), tpb, 0, stream>>>(accb, negb, catb);
  float* out2r = gout;                            // gout dead, reuse as GEMM output
  {
    dim3 g(1, M1_PAD / 128);
    gemm_bf16_wmma<<<g, tpb, 0, stream>>>(catb, 128, wgreT, 128, out2r, 64,
                                          M1_ROWS, 64, 128);
  }
  float* out2buf = hlin;                          // hlin dead, reuse as [N,64,10]
  k_relu_out2<<<blocks((long long)M1_ROWS * 64), tpb, 0, stream>>>(out2r, gre_b, out2buf);

  // ---- TCN2: im2col + WMMA GEMM (M2 x 192 x 128) + gated activation -> d_out ----
  unsigned short* A2 = A1;                        // im2col buffer reuse (M2 is tile-aligned)
  k_im2col2<<<blocks((long long)M2_ROWS * K_TCN), tpb, 0, stream>>>(out2buf, A2);
  float* G2 = Gbuf;                               // cat dead, reuse for G2
  {
    dim3 g(1, M2_ROWS / 128);
    gemm_bf16_wmma<<<g, tpb, 0, stream>>>(A2, K_TCN, wcat2, K_TCN, G2, 128,
                                          M2_ROWS, 128, K_TCN);
  }
  k_gated2<<<blocks((long long)M2_ROWS * 64), tpb, 0, stream>>>(G2, tc2_b1, tc2_b2, out);
}
